// new_conv_31645319037406
// MI455X (gfx1250) — matrix operand order repaired and checked by reading, not compiled, not run on hardware
//
#include <hip/hip_runtime.h>
#include <hip/hip_bf16.h>

// Problem constants (from the reference)
#define B_     32
#define NB     5
#define CIN    128
#define COUT   128
#define HW     56
#define HW2    (HW*HW)          // 3136
#define KTOT   (9*CIN)          // 1152
#define WPB    (COUT*CIN*9)     // 147456 elements per base, also COUT*KTOT

typedef __attribute__((ext_vector_type(16))) _Float16 v16h;
typedef __attribute__((ext_vector_type(8)))  _Float16 h8;
typedef __attribute__((ext_vector_type(8)))  float    v8f;

// ---------------------------------------------------------------------------
// Kernel 1: c[n] = scales[n] * mean(|w_n|)   (one block per base)
// ---------------------------------------------------------------------------
__global__ __launch_bounds__(256)
void k_absmean(const float* __restrict__ w, const float* __restrict__ scales,
               float* __restrict__ c) {
  __shared__ float s[256];
  const int n = blockIdx.x;
  const float* wn = w + (size_t)n * WPB;
  float sum = 0.f;
  for (int i = threadIdx.x; i < WPB; i += 256) sum += fabsf(wn[i]);
  s[threadIdx.x] = sum;
  __syncthreads();
  for (int o = 128; o > 0; o >>= 1) {
    if (threadIdx.x < o) s[threadIdx.x] += s[threadIdx.x + o];
    __syncthreads();
  }
  if (threadIdx.x == 0) c[n] = scales[n] * (s[0] / (float)WPB);
}

// ---------------------------------------------------------------------------
// Kernel 2: W_eff[o][k] = sum_n c[n]*sign(w[n,o,ci,kh,kw]),  k = r*128 + ci,
//           split into f16 hi/lo halves for the 3-product WMMA scheme.
// ---------------------------------------------------------------------------
__global__ __launch_bounds__(256)
void k_buildw(const float* __restrict__ w, const float* __restrict__ c,
              _Float16* __restrict__ Whi, _Float16* __restrict__ Wlo) {
  const int idx = blockIdx.x * 256 + threadIdx.x;   // 0 .. COUT*KTOT-1
  const int o  = idx / KTOT;
  const int k  = idx - o * KTOT;
  const int r  = k >> 7;          // tap index 0..8
  const int ci = k & 127;
  const int base = (o * CIN + ci) * 9 + r;          // [n][o][ci][kh][kw]
  float sum = 0.f;
#pragma unroll
  for (int n = 0; n < NB; ++n) {
    float wv = w[(size_t)n * WPB + base];
    float sg = (wv > 0.f) ? 1.f : ((wv < 0.f) ? -1.f : 0.f);
    sum += c[n] * sg;
  }
  _Float16 hi = (_Float16)sum;
  Whi[idx] = hi;
  Wlo[idx] = (_Float16)(sum - (float)hi);
}

// ---------------------------------------------------------------------------
// Kernel 3: implicit-GEMM conv via v_wmma_f32_16x16x32_f16.
//   Block: one image, 64 consecutive output positions, all 128 out channels.
//   Stage x window (flat range [p0-57, p0+122], all 128 ci) in LDS as f32,
//   then per tap build the 64x128 f16 hi/lo patch tile and run 48 WMMAs.
// ---------------------------------------------------------------------------
#define XS_W      184   // 180 used + pad; 736B row stride (16B aligned)
#define BT_STRIDE 136   // 128 + 8 halfs pad; 272B row stride (16B aligned)

__global__ __launch_bounds__(256)
void k_conv_wmma(const float* __restrict__ x, const _Float16* __restrict__ Whi,
                 const _Float16* __restrict__ Wlo, float* __restrict__ out) {
  __shared__ __align__(16) float    xs[CIN * XS_W];          // 94208 B
  __shared__ __align__(16) _Float16 bt_hi[64 * BT_STRIDE];   // 17408 B
  __shared__ __align__(16) _Float16 bt_lo[64 * BT_STRIDE];   // 17408 B

  const int tid  = threadIdx.x;
  const int lane = tid & 31;
  const int wid  = tid >> 5;            // 8 waves
  const int g    = lane >> 4;           // half-wave group
  const int ln   = lane & 15;

  const int b  = blockIdx.x / 49;       // 3136/64 = 49 N-tiles per image
  const int p0 = (blockIdx.x - b * 49) * 64;

  const int wave_m = (wid >> 1) * 32;   // 4 wave-rows  -> M tile 128
  const int wave_n = (wid & 1) * 32;    // 2 wave-cols  -> N tile 64

  // ---- stage x window into LDS (f32): xs[ci][j], j <-> flat p0-57+j -------
  {
    const int jlane = tid & 63;
    const int ci0   = tid >> 6;         // 0..3
    const int qbase = p0 - 57;
    for (int ci = ci0; ci < CIN; ci += 4) {
      const float* xp = x + ((size_t)b * CIN + ci) * HW2 + qbase;
      float* xr = xs + ci * XS_W;
      for (int j = jlane; j < 180; j += 64) {
        const int q = qbase + j;
        xr[j] = (q >= 0 && q < HW2) ? xp[j] : 0.f;
      }
    }
  }

  // ---- per-thread invariants for the patch build --------------------------
  const int bn  = tid & 63;             // patch row (spatial position)
  const int bc0 = tid >> 6;             // starting channel 0..3 (stride 4)
  const int p   = p0 + bn;
  const int oh  = p / HW;
  const int ow  = p - oh * HW;

  // ---- per-thread invariant A-fragment row pointers (weights in L2) -------
  const _Float16* wh_row[2];
  const _Float16* wl_row[2];
#pragma unroll
  for (int mi = 0; mi < 2; ++mi) {
    const int row = wave_m + 16 * mi + ln;
    wh_row[mi] = Whi + (size_t)row * KTOT + 8 * g;
    wl_row[mi] = Wlo + (size_t)row * KTOT + 8 * g;
  }

  v8f acc[2][2] = {};

  for (int r = 0; r < 9; ++r) {
    const int dh = r / 3 - 1;
    const int dw = r % 3 - 1;

    __syncthreads();   // previous tap's fragment reads done before rewrite
    // ---- build 64x128 patch tile from the LDS f32 stage ----
    {
      const int ih = oh + dh, iw = ow + dw;
      const bool inb = (ih >= 0) & (ih < HW) & (iw >= 0) & (iw < HW);
      const int j = bn + 57 + dh * HW + dw;          // stage column
      const float* xr = xs + bc0 * XS_W + j;
      _Float16* bh = bt_hi + bn * BT_STRIDE + bc0;
      _Float16* bl = bt_lo + bn * BT_STRIDE + bc0;
#pragma unroll 8
      for (int i = 0; i < 32; ++i) {                 // ci = bc0 + 4*i
        const float v = inb ? xr[i * 4 * XS_W] : 1.0f;
        const _Float16 h = (_Float16)v;
        bh[i * 4] = h;
        bl[i * 4] = (_Float16)(v - (float)h);
      }
    }
    __syncthreads();

#pragma unroll
    for (int cc = 0; cc < 4; ++cc) {
      const int kb = r * 128 + cc * 32;

      // A fragments: lane ln holds row M; halfs K[8g..+7] then K[16+8g..+7]
      v16h ahi[2], alo[2];
#pragma unroll
      for (int mi = 0; mi < 2; ++mi) {
        const h8* ph = (const h8*)(wh_row[mi] + kb);
        const h8* pl = (const h8*)(wl_row[mi] + kb);
        h8 h0 = ph[0], h1 = ph[2];   // +2 h8 = +16 halfs
        h8 l0 = pl[0], l1 = pl[2];
#pragma unroll
        for (int jj = 0; jj < 8; ++jj) {
          ahi[mi][jj] = h0[jj]; ahi[mi][8 + jj] = h1[jj];
          alo[mi][jj] = l0[jj]; alo[mi][8 + jj] = l1[jj];
        }
      }

      // B fragments: lane ln = column N; halfs K[8g..+7] then K[16+8g..+7]
      v16h bhi[2], blo[2];
#pragma unroll
      for (int ni = 0; ni < 2; ++ni) {
        const int ncol = wave_n + 16 * ni + ln;
        const h8* ph = (const h8*)(bt_hi + ncol * BT_STRIDE + cc * 32 + 8 * g);
        const h8* pl = (const h8*)(bt_lo + ncol * BT_STRIDE + cc * 32 + 8 * g);
        h8 h0 = ph[0], h1 = ph[2];
        h8 l0 = pl[0], l1 = pl[2];
#pragma unroll
        for (int jj = 0; jj < 8; ++jj) {
          bhi[ni][jj] = h0[jj]; bhi[ni][8 + jj] = h1[jj];
          blo[ni][jj] = l0[jj]; blo[ni][8 + jj] = l1[jj];
        }
      }

      // 3-product f16 split: hi*hi + hi*lo + lo*hi (f32 accumulate)
#pragma unroll
      for (int mi = 0; mi < 2; ++mi)
#pragma unroll
        for (int ni = 0; ni < 2; ++ni) {
          acc[mi][ni] = __builtin_amdgcn_wmma_f32_16x16x32_f16(
              false, ahi[mi], false, bhi[ni], (short)0, acc[mi][ni], false, false);
          acc[mi][ni] = __builtin_amdgcn_wmma_f32_16x16x32_f16(
              false, ahi[mi], false, blo[ni], (short)0, acc[mi][ni], false, false);
          acc[mi][ni] = __builtin_amdgcn_wmma_f32_16x16x32_f16(
              false, alo[mi], false, bhi[ni], (short)0, acc[mi][ni], false, false);
        }
    }
  }

  // ---- epilogue: D layout -> NCHW f32 output ----
#pragma unroll
  for (int mi = 0; mi < 2; ++mi)
#pragma unroll
    for (int ni = 0; ni < 2; ++ni) {
      const int pp = p0 + wave_n + 16 * ni + ln;
      const int mb = wave_m + 16 * mi + 8 * g;     // M = mb + v
      float* op = out + ((size_t)b * COUT + mb) * HW2 + pp;
      v8f a = acc[mi][ni];
#pragma unroll
      for (int v = 0; v < 8; ++v) op[(size_t)v * HW2] = a[v];
    }
}

// ---------------------------------------------------------------------------
extern "C" void kernel_launch(void* const* d_in, const int* in_sizes, int n_in,
                              void* d_out, int out_size, void* d_ws, size_t ws_size,
                              hipStream_t stream) {
  const float* x      = (const float*)d_in[0];   // [32,128,56,56]
  const float* w      = (const float*)d_in[1];   // [5,128,128,3,3]
  const float* scales = (const float*)d_in[2];   // [5]
  float* out = (float*)d_out;                    // [32,128,56,56]

  _Float16* Whi = (_Float16*)d_ws;               // COUT*KTOT halfs
  _Float16* Wlo = Whi + (size_t)COUT * KTOT;
  float*    c   = (float*)(Wlo + (size_t)COUT * KTOT);  // 5 floats

  k_absmean<<<NB, 256, 0, stream>>>(w, scales, c);
  k_buildw<<<(COUT * KTOT) / 256, 256, 0, stream>>>(w, c, Whi, Wlo);
  k_conv_wmma<<<B_ * (HW2 / 64), 256, 0, stream>>>(x, Whi, Wlo, out);
}
